// AttentionTrain_89283780149596
// MI455X (gfx1250) — matrix operand order repaired and checked by reading, not compiled, not run on hardware
//
#include <hip/hip_runtime.h>
#include <hip/hip_bf16.h>
#include <math.h>

typedef unsigned short u16;
typedef unsigned int   u32;
typedef unsigned long long u64;
typedef __attribute__((ext_vector_type(16))) __bf16 v16bf;
typedef __attribute__((ext_vector_type(8)))  float  v8f;

#define DM    1024
#define HEADS 16
#define HD    64
#define BATCH 4
#define SEQ   2048
#define MS    (BATCH*SEQ)   /* 8192 rows */

__device__ __forceinline__ u16 f2bf(float f) {
  u32 u = __float_as_uint(f);
  u32 r = (u + 0x7FFFu + ((u >> 16) & 1u)) >> 16;   // round-to-nearest-even
  return (u16)r;
}

union Frag16 { v16bf bf; u32 u[8]; };

__device__ __forceinline__ v8f wmma_bf16(const Frag16& a, const Frag16& b, v8f c) {
  return __builtin_amdgcn_wmma_f32_16x16x32_bf16(false, a.bf, false, b.bf,
                                                 (short)0, c, false, false);
}

// 16-lane row butterfly via DPP only (no LDS): xor1, xor2, half_mirror(xor7), mirror(xor15)
template<int CTRL>
__device__ __forceinline__ float dpp_movf(float x) {
  int i = __float_as_int(x);
  return __int_as_float(__builtin_amdgcn_update_dpp(i, i, CTRL, 0xF, 0xF, true));
}
__device__ __forceinline__ float row_max16(float x) {
  x = fmaxf(x, dpp_movf<0xB1>(x));   // quad_perm [1,0,3,2]  (xor 1)
  x = fmaxf(x, dpp_movf<0x4E>(x));   // quad_perm [2,3,0,1]  (xor 2)
  x = fmaxf(x, dpp_movf<0x141>(x));  // row_half_mirror      (xor 7)
  x = fmaxf(x, dpp_movf<0x140>(x));  // row_mirror           (xor 15)
  return x;
}
__device__ __forceinline__ float row_sum16(float x) {
  x += dpp_movf<0xB1>(x);
  x += dpp_movf<0x4E>(x);
  x += dpp_movf<0x141>(x);
  x += dpp_movf<0x140>(x);
  return x;
}

// ---------------------------------------------------------------- cast f32->bf16 (with scale)
__global__ __launch_bounds__(256)
void cast_f32_bf16_kernel(const float* __restrict__ in, u16* __restrict__ out,
                          int n4, float scale) {
  int i = blockIdx.x * blockDim.x + threadIdx.x;
  if (i < n4) {
    float4 v = ((const float4*)in)[i];
    u32 lo = (u32)f2bf(v.x*scale) | ((u32)f2bf(v.y*scale) << 16);
    u32 hi = (u32)f2bf(v.z*scale) | ((u32)f2bf(v.w*scale) << 16);
    ((uint2*)out)[i] = make_uint2(lo, hi);
  }
}

// ---------------------------------------------------------------- bf16 GEMM + bias
// C[M,N] = A[M,K] @ W[K,N] + bias*bscale ; 128x128 block, 8 waves, wave 32x64, BK=64.
// Double-buffered LDS: A tile via global_load_async_to_lds_b128 (ASYNCcnt) for tile t+1
// issued before computing tile t; W tile loaded to VGPRs early, scattered (transposed)
// into LDS after compute; single barrier per K-tile.
template<bool OUT_F32>
__global__ __launch_bounds__(256)
void gemm_bf16_kernel(const u16* __restrict__ A, const u16* __restrict__ W,
                      const float* __restrict__ bias, float bscale,
                      void* __restrict__ Cout, int M, int N, int K)
{
  __shared__ __align__(16) u16 Asm[2][128*64];    // [m][k]  2 x 16 KiB
  __shared__ __align__(16) u16 BsmT[2][128*64];   // [n][k]  2 x 16 KiB

  const int tid  = threadIdx.x;
  const int lane = tid & 31, wave = tid >> 5;
  const int wm = wave & 3, wn = wave >> 2;        // 4x2 wave grid
  const int half = lane >> 4, l16 = lane & 15;
  const int mb = blockIdx.y * 128, nb = blockIdx.x * 128;
  const u64 abase = (u64)(size_t)A;
  const int sArow = tid >> 1, sAch = (tid & 1) * 32;   // A staging map
  const int sBkk  = tid >> 2, sBn0 = (tid & 3) * 32;   // B staging map

  const v8f vzero = {0.f,0.f,0.f,0.f,0.f,0.f,0.f,0.f};
  v8f acc[2][4];
  for (int i = 0; i < 2; i++) for (int j = 0; j < 4; j++) acc[i][j] = vzero;

  // ---- staging helpers
  auto stageA_async = [&](int kb, int buf) {
    u32 lbase = (u32)(size_t)&Asm[buf][sArow*64 + sAch];
    u32 gbyte = ((u32)(mb+sArow)*(u32)K + (u32)(kb + sAch)) * 2u;
    #pragma unroll
    for (int j = 0; j < 4; j++)
      asm volatile("global_load_async_to_lds_b128 %0, %1, %2"
                   :: "v"(lbase + 16u*j), "v"(gbyte + 16u*j), "s"(abase) : "memory");
  };
  auto loadB = [&](int kb, uint4* wpre) {
    const uint4* g = (const uint4*)&W[(size_t)(kb+sBkk)*N + nb + sBn0];
    #pragma unroll
    for (int q = 0; q < 4; q++) wpre[q] = g[q];
  };
  auto scatterB = [&](int buf, const uint4* wpre) {
    #pragma unroll
    for (int q = 0; q < 4; q++) {
      const u16* wsp = (const u16*)&wpre[q];
      #pragma unroll
      for (int j = 0; j < 8; j++) BsmT[buf][(sBn0 + q*8 + j)*64 + sBkk] = wsp[j];
    }
  };

  // ---- prologue: stage tile 0
  {
    uint4 wpre[4];
    stageA_async(0, 0);
    loadB(0, wpre);
    scatterB(0, wpre);
    asm volatile("s_wait_asynccnt 0x0" ::: "memory");
    __syncthreads();
  }

  const int nk = K / 64;
  for (int t = 0; t < nk; t++) {
    const int kb = t * 64, buf = t & 1;
    uint4 wpre[4];
    const bool more = (t + 1 < nk);
    if (more) {
      stageA_async(kb + 64, buf ^ 1);       // ASYNCcnt rides under compute
      loadB(kb + 64, wpre);                 // global latency hidden by WMMAs
      if (t + 2 < nk)
        __builtin_prefetch(&W[(size_t)(kb + 128 + sBkk)*N + nb + sBn0], 0, 1);
    }

    #pragma unroll
    for (int ks = 0; ks < 2; ks++) {
      Frag16 af[2], bfr[4];
      #pragma unroll
      for (int mt = 0; mt < 2; mt++) {
        int row = wm*32 + mt*16 + l16;
        #pragma unroll
        for (int r = 0; r < 8; r++) {
          int k = ks*32 + 16*(r>>2) + 8*half + 2*(r&3);   // A-frag 16x32 layout
          af[mt].u[r] = *(const u32*)&Asm[buf][row*64 + k];
        }
      }
      #pragma unroll
      for (int nt = 0; nt < 4; nt++) {
        int col = wn*64 + nt*16 + l16;
        #pragma unroll
        for (int r = 0; r < 8; r++) {
          int k = ks*32 + 16*(r>>2) + 8*half + 2*(r&3);   // B-frag 32x16 layout
          bfr[nt].u[r] = *(const u32*)&BsmT[buf][col*64 + k];
        }
      }
      #pragma unroll
      for (int mt = 0; mt < 2; mt++)
        #pragma unroll
        for (int nt = 0; nt < 4; nt++)
          acc[mt][nt] = wmma_bf16(af[mt], bfr[nt], acc[mt][nt]);
    }

    if (more) scatterB(buf ^ 1, wpre);      // after compute: keeps DS pipe clean
    asm volatile("s_wait_asynccnt 0x0" ::: "memory");
    __syncthreads();
  }

  #pragma unroll
  for (int mt = 0; mt < 2; mt++)
    #pragma unroll
    for (int nt = 0; nt < 4; nt++) {
      int col = nb + wn*64 + nt*16 + l16;
      float bv = bias[col] * bscale;
      #pragma unroll
      for (int r = 0; r < 8; r++) {
        int row = mb + wm*32 + mt*16 + r + 8*half;        // D-layout: m = r + 8*half
        float v = acc[mt][nt][r] + bv;
        if (OUT_F32) ((float*)Cout)[(size_t)row*N + col] = v;
        else         ((u16*)  Cout)[(size_t)row*N + col] = f2bf(v);
      }
    }
}

// ---------------------------------------------------------------- flash attention
// Q pre-scaled by 1/sqrt(DM) (folded into Wq cast). Double-buffered K/V tiles:
// K staged async, V loaded to VGPRs early and scattered (transposed) after compute.
__global__ __launch_bounds__(256)
void attn_kernel(const u16* __restrict__ Q, const u16* __restrict__ K,
                 const u16* __restrict__ V, u16* __restrict__ ctx)
{
  __shared__ __align__(16) u16 Ksm[2][32*64];     // [key][dim]  2 x 4 KiB
  __shared__ __align__(16) u16 VsmT[2][64*32];    // [dim][key]  2 x 4 KiB
  __shared__ __align__(16) u16 Psm[8][16*32];     // per-wave P scratch, 8 KiB

  const int tid  = threadIdx.x;
  const int lane = tid & 31, wave = tid >> 5;
  const int half = lane >> 4, l16 = lane & 15;
  const int b = blockIdx.y >> 4, h = blockIdx.y & 15;
  const int qbase = blockIdx.x * 128;
  const int qw = qbase + wave * 16;               // this wave's 16 q rows
  const size_t baseRow = (size_t)b * SEQ;
  const u64 kbase = (u64)(size_t)(K + baseRow*DM + (size_t)h*HD);   // uniform
  const int sKey = tid >> 3, sD0 = (tid & 7) * 8; // staging map

  // Q fragments (dims 0..31 and 32..63) resident in registers
  Frag16 qf[2];
  {
    size_t qoff = (baseRow + qw + l16) * DM + (size_t)h * HD;
    #pragma unroll
    for (int dc = 0; dc < 2; dc++)
      #pragma unroll
      for (int r = 0; r < 8; r++) {
        int k = dc*32 + 16*(r>>2) + 8*half + 2*(r&3);
        qf[dc].u[r] = *(const u32*)&Q[qoff + k];
      }
  }

  const v8f vzero = {0.f,0.f,0.f,0.f,0.f,0.f,0.f,0.f};
  v8f o[4]; for (int i = 0; i < 4; i++) o[i] = vzero;
  float mrow[8], lrow[8];
  #pragma unroll
  for (int r = 0; r < 8; r++) { mrow[r] = -__builtin_inff(); lrow[r] = 0.f; }

  const int ntiles = qbase/32 + 4;                // causal: keys up to qbase+127

  auto stageK_async = [&](int kb, int buf) {
    u32 lk = (u32)(size_t)&Ksm[buf][sKey*64 + sD0];
    u32 go = ((u32)(kb + sKey) * (u32)DM + (u32)sD0) * 2u;
    asm volatile("global_load_async_to_lds_b128 %0, %1, %2"
                 :: "v"(lk), "v"(go), "s"(kbase) : "memory");
  };
  auto loadV = [&](int kb) {
    return *(const uint4*)&V[(baseRow + kb + sKey)*DM + (size_t)h*HD + sD0];
  };
  auto scatterV = [&](int buf, uint4 vvec) {
    const u16* vs = (const u16*)&vvec;
    #pragma unroll
    for (int j = 0; j < 8; j++) VsmT[buf][(sD0+j)*32 + sKey] = vs[j];
  };

  // ---- prologue: stage tile 0
  {
    stageK_async(0, 0);
    scatterV(0, loadV(0));
    asm volatile("s_wait_asynccnt 0x0" ::: "memory");
    __syncthreads();
  }

  for (int jt = 0; jt < ntiles; jt++) {
    const int kb = jt * 32, buf = jt & 1;
    uint4 vpre;
    const bool more = (jt + 1 < ntiles);
    if (more) {
      stageK_async(kb + 32, buf ^ 1);
      vpre = loadV(kb + 32);
      if (jt + 2 < ntiles)
        __builtin_prefetch(&V[(baseRow + kb + 64 + sKey)*DM + (size_t)h*HD + sD0], 0, 1);
    }

    if (kb <= qw + 15) {                          // wave-uniform; EXEC stays full
      // ---- scores S[16x32] = Q . K^T (Q pre-scaled)
      v8f s0 = vzero, s1 = vzero;
      #pragma unroll
      for (int dc = 0; dc < 2; dc++) {
        Frag16 bk0, bk1;
        #pragma unroll
        for (int r = 0; r < 8; r++) {
          int k = dc*32 + 16*(r>>2) + 8*half + 2*(r&3);
          bk0.u[r] = *(const u32*)&Ksm[buf][l16*64 + k];        // keys 0..15
          bk1.u[r] = *(const u32*)&Ksm[buf][(16+l16)*64 + k];   // keys 16..31
        }
        s0 = wmma_bf16(qf[dc], bk0, s0);
        s1 = wmma_bf16(qf[dc], bk1, s1);
      }
      // ---- online softmax; row m = r + 8*half, col n = l16 (D-layout)
      const bool needMask = (kb + 31 > qw);       // wave-uniform: diagonal-band tile
      float p0[8], p1[8], sc[8];
      #pragma unroll
      for (int r = 0; r < 8; r++) {
        float a0, a1;
        if (needMask) {
          int qrow = qw + r + 8*half;
          a0 = (kb + l16      <= qrow) ? s0[r] : -__builtin_inff();
          a1 = (kb + 16 + l16 <= qrow) ? s1[r] : -__builtin_inff();
        } else { a0 = s0[r]; a1 = s1[r]; }
        float rm = row_max16(fmaxf(a0, a1));
        float mn = fmaxf(mrow[r], rm);
        sc[r] = __expf(mrow[r] - mn);
        p0[r] = __expf(a0 - mn);
        p1[r] = __expf(a1 - mn);
        float rs = row_sum16(p0[r] + p1[r]);
        lrow[r] = lrow[r]*sc[r] + rs;
        mrow[r] = mn;
      }
      #pragma unroll
      for (int nt = 0; nt < 4; nt++)
        #pragma unroll
        for (int r = 0; r < 8; r++) o[nt][r] *= sc[r];

      // ---- P: D-layout -> A-fragment via per-wave LDS round-trip
      #pragma unroll
      for (int r = 0; r < 8; r++) {
        int m = r + 8*half;
        Psm[wave][m*32 + l16]      = f2bf(p0[r]);
        Psm[wave][m*32 + 16 + l16] = f2bf(p1[r]);
      }
      asm volatile("s_wait_dscnt 0x0" ::: "memory");
      Frag16 pf;
      #pragma unroll
      for (int r = 0; r < 8; r++) {
        int k = 16*(r>>2) + 8*half + 2*(r&3);
        pf.u[r] = *(const u32*)&Psm[wave][l16*32 + k];
      }
      // ---- O += P[16x32] . V[32x64]
      #pragma unroll
      for (int nt = 0; nt < 4; nt++) {
        Frag16 vf;
        #pragma unroll
        for (int r = 0; r < 8; r++) {
          int kk = 16*(r>>2) + 8*half + 2*(r&3);
          vf.u[r] = *(const u32*)&VsmT[buf][(nt*16 + l16)*32 + kk];
        }
        o[nt] = wmma_bf16(pf, vf, o[nt]);
      }
    }

    if (more) scatterV(buf ^ 1, vpre);            // after compute: DS pipe stays clean
    asm volatile("s_wait_asynccnt 0x0" ::: "memory");
    __syncthreads();
  }

  // ---- normalize and write ctx (bf16) in [b, s, h*d] layout
  #pragma unroll
  for (int r = 0; r < 8; r++) {
    float inv = 1.0f / lrow[r];
    size_t row = baseRow + qw + r + 8*half;
    #pragma unroll
    for (int nt = 0; nt < 4; nt++)
      ctx[row*DM + (size_t)h*HD + nt*16 + l16] = f2bf(o[nt][r] * inv);
  }
}

// ---------------------------------------------------------------- launch
extern "C" void kernel_launch(void* const* d_in, const int* in_sizes, int n_in,
                              void* d_out, int out_size, void* d_ws, size_t ws_size,
                              hipStream_t stream) {
  (void)in_sizes; (void)n_in; (void)out_size; (void)ws_size;
  const float* x  = (const float*)d_in[0];
  const float* Wq = (const float*)d_in[1];
  const float* bq = (const float*)d_in[2];
  const float* Wk = (const float*)d_in[3];
  const float* bk = (const float*)d_in[4];
  const float* Wv = (const float*)d_in[5];
  const float* bv = (const float*)d_in[6];
  const float* Wo = (const float*)d_in[7];
  const float* bo = (const float*)d_in[8];
  float* out = (float*)d_out;

  char* ws = (char*)d_ws;
  const size_t actB = (size_t)MS * DM * sizeof(u16);  // 16 MiB
  const size_t wB   = (size_t)DM * DM * sizeof(u16);  //  2 MiB
  u16* xb   = (u16*)ws; ws += actB;
  u16* wqb  = (u16*)ws; ws += wB;
  u16* wkb  = (u16*)ws; ws += wB;
  u16* wvb  = (u16*)ws; ws += wB;
  u16* wob  = (u16*)ws; ws += wB;
  u16* Qb   = (u16*)ws; ws += actB;
  u16* Kb   = (u16*)ws; ws += actB;
  u16* Vb   = (u16*)ws; ws += actB;
  u16* ctxb = (u16*)ws; ws += actB;

  const float qsc = 0.03125f;  // 1/sqrt(1024), folded into Wq (and bq via bscale)
  const int n4a = MS*DM/4, n4w = DM*DM/4;
  cast_f32_bf16_kernel<<<(n4a+255)/256, 256, 0, stream>>>(x,  xb,  n4a, 1.0f);
  cast_f32_bf16_kernel<<<(n4w+255)/256, 256, 0, stream>>>(Wq, wqb, n4w, qsc);
  cast_f32_bf16_kernel<<<(n4w+255)/256, 256, 0, stream>>>(Wk, wkb, n4w, 1.0f);
  cast_f32_bf16_kernel<<<(n4w+255)/256, 256, 0, stream>>>(Wv, wvb, n4w, 1.0f);
  cast_f32_bf16_kernel<<<(n4w+255)/256, 256, 0, stream>>>(Wo, wob, n4w, 1.0f);

  dim3 gg(DM/128, MS/128);   // (8, 64)
  gemm_bf16_kernel<false><<<gg, 256, 0, stream>>>(xb, wqb, bq, qsc,  Qb, MS, DM, DM);
  gemm_bf16_kernel<false><<<gg, 256, 0, stream>>>(xb, wkb, bk, 1.0f, Kb, MS, DM, DM);
  gemm_bf16_kernel<false><<<gg, 256, 0, stream>>>(xb, wvb, bv, 1.0f, Vb, MS, DM, DM);

  attn_kernel<<<dim3(SEQ/128, BATCH*HEADS), 256, 0, stream>>>(Qb, Kb, Vb, ctxb);

  gemm_bf16_kernel<true><<<gg, 256, 0, stream>>>(ctxb, wob, bo, 1.0f, out, MS, DM, DM);
}
